// CapsuleLayer_94489281200
// MI455X (gfx1250) — matrix operand order repaired and checked by reading, not compiled, not run on hardware
//
#include <hip/hip_runtime.h>

typedef __attribute__((ext_vector_type(16))) _Float16 v16h;
typedef __attribute__((ext_vector_type(8)))  float    v8f;

#define R_DIM 2048
#define C_DIM 16
#define B_DIM 32
#define O_DIM 32
#define K_DIM 32

union AFrag { v16h v; _Float16 h[16]; };

// ---------------------------------------------------------------------------
// CDNA5 async global->LDS copy (ASYNCcnt-tracked), guarded so the file
// compiles on toolchains without the builtin (falls back to load+ds_store).
// Probe round 2: builtin exists; signature wants (AS1 int*, AS3 int*, imm, imm).
// ---------------------------------------------------------------------------
#if defined(__has_builtin)
#  if __has_builtin(__builtin_amdgcn_global_load_async_to_lds_b32)
#    define HAVE_ASYNC_LDS 1
#  endif
#endif
#ifndef HAVE_ASYNC_LDS
#  define HAVE_ASYNC_LDS 0
#endif

typedef __attribute__((address_space(1))) int gint_t;   // global
typedef __attribute__((address_space(3))) int lint_t;   // LDS

__device__ __forceinline__ void cp_b32(const float* g, float* l) {
#if HAVE_ASYNC_LDS
    __builtin_amdgcn_global_load_async_to_lds_b32((gint_t*)g, (lint_t*)l, 0, 0);
#else
    *l = *g;
#endif
}
// wait until <=16 outstanding (previous tile retired; in-order per wave)
__device__ __forceinline__ void wait_async_prev() {
#if HAVE_ASYNC_LDS
#  if defined(__has_builtin) && __has_builtin(__builtin_amdgcn_s_wait_asynccnt)
    __builtin_amdgcn_s_wait_asynccnt(16);
#  else
    asm volatile("s_wait_asynccnt 0x10" ::: "memory");
#  endif
#endif
}
__device__ __forceinline__ void wait_async_all() {
#if HAVE_ASYNC_LDS
#  if defined(__has_builtin) && __has_builtin(__builtin_amdgcn_s_wait_asynccnt)
    __builtin_amdgcn_s_wait_asynccnt(0);
#  else
    asm volatile("s_wait_asynccnt 0x0" ::: "memory");
#  endif
#endif
}

// ---------------------------------------------------------------------------
// init: probs = 1/R (iteration 0 softmax of zeros), S = 0 (atomic accumulator)
// ---------------------------------------------------------------------------
__global__ void cap_init(float* __restrict__ probs, float* __restrict__ S) {
    int i = blockIdx.x * 256 + threadIdx.x;
    if (i < K_DIM * R_DIM * B_DIM) probs[i] = 1.0f / (float)R_DIM;
    if (i < K_DIM * B_DIM * O_DIM) S[i] = 0.0f;
}

// ---------------------------------------------------------------------------
// S[k,b,o] += sum_{rc} (probs[k,r,b]*x[b,rc]) * w[k][rc,o]
// batched GEMM per k: M=b(32), N=o(32), K=rc(32768), WMMA f16 16x16x32
// grid: 32 k * 8 r-blocks = 256 blocks, 256 threads (8 waves)
// wave -> (D-tile tb,to) x (rc-half h); 4 r (64 rc) per tile, double-buffered
// ---------------------------------------------------------------------------
__global__ __launch_bounds__(256) void cap_gemmS(
    const float* __restrict__ x, const float* __restrict__ w,
    const float* __restrict__ probs, float* __restrict__ S)
{
    __shared__ float lX[2][32 * 68];   // [b][rc_local 0..63], padded stride
    __shared__ float lW[2][64 * 34];   // [rc_local][o], padded stride
    __shared__ float lP[2][128];       // [b][r_in_4]
    __shared__ float lRed[4 * 32 * 8]; // cross-wave tile reduction

    const int tid  = threadIdx.x;
    const int lane = tid & 31;
    const int wave = tid >> 5;
    const int tileid = wave & 3, h = wave >> 2;
    const int tb = tileid >> 1, to = tileid & 1;

    const int k    = blockIdx.x >> 3;
    const int rblk = blockIdx.x & 7;
    const int r0   = rblk * 256;

    const float* wk = w + (size_t)k * (R_DIM * C_DIM * O_DIM);

    // stage one 4-r tile (16 async b32 per thread -> LDS buffer `buf`)
    auto stage = [&](int buf, int it) {
        const int rIter  = r0 + it * 4;
        const int rcBase = rIter * C_DIM;
        const float* wsrc = wk + (size_t)rcBase * O_DIM;   // contiguous 8 KiB
        __builtin_prefetch(wsrc + 2048, 0, 1);             // next panel -> L2
#pragma unroll
        for (int i = 0; i < 8; ++i) {                      // W [64][32]
            int q = tid + 256 * i;
            cp_b32(wsrc + q, &lW[buf][(q >> 5) * 34 + (q & 31)]);
        }
#pragma unroll
        for (int i = 0; i < 8; ++i) {                      // X [32][64]
            int q = tid + 256 * i;
            int b = q >> 6, j = q & 63;
            cp_b32(x + b * (R_DIM * C_DIM) + rcBase + j, &lX[buf][b * 68 + j]);
        }
        if (tid < 128) {                                   // probs [32][4]
            int rr = tid >> 5, b = tid & 31;
            lP[buf][b * 4 + rr] =
                probs[((size_t)k * R_DIM + rIter + rr) * B_DIM + b];
        }
    };

    v8f acc;
#pragma unroll
    for (int i = 0; i < 8; ++i) acc[i] = 0.0f;

    const int m    = lane & 15;
    const int kb   = (lane < 16) ? 0 : 8;
    const int bRow = tb * 16 + m;
    const int oCol = to * 16 + (lane & 15);

    stage(0, 0);
    for (int it = 0; it < 64; ++it) {
        const int buf = it & 1;
        if (it + 1 < 64) { stage(buf ^ 1, it + 1); wait_async_prev(); }
        else             { wait_async_all(); }
        __syncthreads();   // tile `it` visible to all waves

        // A = (probs * x)  16x32 f16 ; B = w  32x16 f16 ; scale BEFORE cvt
        AFrag a, bf;
        const float s0 = lP[buf][bRow * 4 + 2 * h];
        const float s1 = lP[buf][bRow * 4 + 2 * h + 1];
#pragma unroll
        for (int j = 0; j < 16; ++j) {
            int K   = kb + ((j < 8) ? j : j + 8);
            int rcl = h * 32 + K;
            a.h[j]  = (_Float16)(lX[buf][bRow * 68 + rcl] * ((j < 8) ? s0 : s1));
            int Ki  = kb + ((j < 8) ? j : j + 8);
            bf.h[j] = (_Float16)lW[buf][(h * 32 + Ki) * 34 + oCol];
        }
        acc = __builtin_amdgcn_wmma_f32_16x16x32_f16(false, a.v, false, bf.v,
                                                     (short)0, acc, false, false);
        __syncthreads();   // buffer `buf` free for re-staging next iteration
    }

    // reduce the two rc-halves, then atomically fold r-block partials into S
    if (h == 1) {
#pragma unroll
        for (int i = 0; i < 8; ++i) lRed[(tileid * 32 + lane) * 8 + i] = acc[i];
    }
    __syncthreads();
    if (h == 0) {
#pragma unroll
        for (int i = 0; i < 8; ++i) {
            float vv = acc[i] + lRed[(tileid * 32 + lane) * 8 + i];
            int bb = tb * 16 + ((lane < 16) ? i : i + 8);
            atomicAdd(&S[(k * B_DIM + bb) * O_DIM + oCol], vv);
        }
    }
}

// ---------------------------------------------------------------------------
// v = sigmoid(S); re-zero S for the next accumulation pass
// ---------------------------------------------------------------------------
__global__ void cap_sigmoid_v(float* __restrict__ S, float* __restrict__ v) {
    int i = blockIdx.x * 256 + threadIdx.x;
    if (i < K_DIM * B_DIM * O_DIM) {
        v[i] = 1.0f / (1.0f + __expf(-S[i]));
        S[i] = 0.0f;
    }
}

// ---------------------------------------------------------------------------
// logits[k,r,b] (+)= sum_c x[b,r,c] * (W[k,r](16x32) @ V[k](32xB))[c,b]
// per (k,r): WMMA M=c(16), N=b(16x2), K=o(32).  B-fragment (V) loop-invariant.
// grid: 256 blocks (k x r-block), 8 waves -> (rsub 0..3) x (bhalf 0..1)
// ---------------------------------------------------------------------------
__global__ __launch_bounds__(256) void cap_logits(
    const float* __restrict__ x, const float* __restrict__ w,
    const float* __restrict__ v, float* __restrict__ logits, int addOld)
{
    __shared__ float lV[32 * 33];           // [b][o]
    __shared__ float lWr[2][64 * 33];       // [(r*16+c)][o] for 4 r
    __shared__ float lXd[2][32 * 4 * 17];   // [(b*4+r)][c]

    const int tid = threadIdx.x, lane = tid & 31, wave = tid >> 5;
    const int rsub = wave >> 1, bhalf = wave & 1;
    const int k = blockIdx.x >> 3, rblk = blockIdx.x & 7;
    const int r0 = rblk * 256;
    const float* wk = w + (size_t)k * (R_DIM * C_DIM * O_DIM);

    auto stage = [&](int buf, int it) {
        const int rIter = r0 + it * 4;
        const float* wsrc = wk + (size_t)rIter * (C_DIM * O_DIM); // 8 KiB
        __builtin_prefetch(wsrc + 2048, 0, 1);
#pragma unroll
        for (int i = 0; i < 8; ++i) {
            int q = tid + 256 * i;
            cp_b32(wsrc + q, &lWr[buf][(q >> 5) * 33 + (q & 31)]);
        }
#pragma unroll
        for (int i = 0; i < 8; ++i) {
            int q = tid + 256 * i;
            int b = q >> 6, rr = (q >> 4) & 3, c = q & 15;
            cp_b32(x + b * (R_DIM * C_DIM) + (rIter + rr) * C_DIM + c,
                   &lXd[buf][(b * 4 + rr) * 17 + c]);
        }
    };

    stage(0, 0);
    for (int q = tid; q < 1024; q += 256)
        lV[(q >> 5) * 33 + (q & 31)] = v[k * 1024 + q];
    __syncthreads();

    // constant B fragment: B[K=o][N=b] = v[k,b,o]
    AFrag bv;
    const int n = lane & 15;
    const int bCol = bhalf * 16 + n;
#pragma unroll
    for (int j = 0; j < 16; ++j) {
        int o = ((lane < 16) ? 0 : 8) + ((j < 8) ? j : j + 8);
        bv.h[j] = (_Float16)lV[bCol * 33 + o];
    }

    const int m  = lane & 15;               // c row
    const int kb = (lane < 16) ? 0 : 8;

    for (int it = 0; it < 64; ++it) {
        const int buf = it & 1;
        const int rIt = r0 + it * 4;
        if (it + 1 < 64) { stage(buf ^ 1, it + 1); wait_async_prev(); }
        else             { wait_async_all(); }
        __syncthreads();

        AFrag aw;                           // A = w[k,r]  16(c) x 32(o)
#pragma unroll
        for (int j = 0; j < 16; ++j) {
            int o = kb + ((j < 8) ? j : j + 8);
            aw.h[j] = (_Float16)lWr[buf][(rsub * 16 + m) * 33 + o];
        }
        v8f d;
#pragma unroll
        for (int i = 0; i < 8; ++i) d[i] = 0.0f;
        d = __builtin_amdgcn_wmma_f32_16x16x32_f16(false, aw.v, false, bv.v,
                                                   (short)0, d, false, false);
        // dot over c with x[b,r,:]  (d[i] <-> c = i or i+8 per lane half)
        float part = 0.0f;
        const int clo = (lane < 16) ? 0 : 8;
#pragma unroll
        for (int i = 0; i < 8; ++i)
            part += lXd[buf][(bCol * 4 + rsub) * 17 + clo + i] * d[i];
        part += __shfl_xor(part, 16, 32);
        if (lane < 16) {
            size_t idx = ((size_t)k * R_DIM + rIt + rsub) * B_DIM + bCol;
            float val = part + (addOld ? logits[idx] : 0.0f);
            logits[idx] = val;              // 16 lanes = 16 consecutive b
        }
        __syncthreads();
    }
}

// ---------------------------------------------------------------------------
// probs[k,r,b] = softmax over r of logits[k,r,b]; one block per (k,b)
// ---------------------------------------------------------------------------
__global__ void cap_softmax(const float* __restrict__ logits,
                            float* __restrict__ probs) {
    const int k = blockIdx.x >> 5, b = blockIdx.x & 31;
    const int tid = threadIdx.x;
    __shared__ float red[256];
    float lv[8];
#pragma unroll
    for (int i = 0; i < 8; ++i)
        lv[i] = logits[((size_t)k * R_DIM + tid + 256 * i) * B_DIM + b];
    float mx = lv[0];
#pragma unroll
    for (int i = 1; i < 8; ++i) mx = fmaxf(mx, lv[i]);
    red[tid] = mx; __syncthreads();
    for (int s = 128; s > 0; s >>= 1) {
        if (tid < s) red[tid] = fmaxf(red[tid], red[tid + s]);
        __syncthreads();
    }
    mx = red[0]; __syncthreads();
    float sum = 0.0f;
#pragma unroll
    for (int i = 0; i < 8; ++i) { lv[i] = __expf(lv[i] - mx); sum += lv[i]; }
    red[tid] = sum; __syncthreads();
    for (int s = 128; s > 0; s >>= 1) {
        if (tid < s) red[tid] += red[tid + s];
        __syncthreads();
    }
    const float inv = 1.0f / red[0];
#pragma unroll
    for (int i = 0; i < 8; ++i)
        probs[((size_t)k * R_DIM + tid + 256 * i) * B_DIM + b] = lv[i] * inv;
}

// ---------------------------------------------------------------------------
// out[k,b,1,1,o] = sigmoid(S2)
// ---------------------------------------------------------------------------
__global__ void cap_out(const float* __restrict__ S, float* __restrict__ out) {
    int i = blockIdx.x * 256 + threadIdx.x;
    if (i < K_DIM * B_DIM * O_DIM) out[i] = 1.0f / (1.0f + __expf(-S[i]));
}

// ---------------------------------------------------------------------------
extern "C" void kernel_launch(void* const* d_in, const int* in_sizes, int n_in,
                              void* d_out, int out_size, void* d_ws, size_t ws_size,
                              hipStream_t stream) {
    const float* x = (const float*)d_in[0];          // [32,2048,16]
    const float* w = (const float*)d_in[1];          // [32,2048,16,32]
    float* out = (float*)d_out;                      // [32,32,1,1,32]
    float* ws  = (float*)d_ws;                       // needs ~16.3 MiB

    float* probs  = ws;                              // [K][R][B] 2,097,152
    float* logits = ws + 2097152;                    // [K][R][B] 2,097,152
    float* S      = ws + 4194304;                    // [K][B][O] 32768
    float* v      = ws + 4227072;                    // [K][B][O] 32768

    dim3 blk(256);
    cap_init     <<<8192, blk, 0, stream>>>(probs, S);
    // iter 0
    cap_gemmS    <<<256,  blk, 0, stream>>>(x, w, probs, S);   // S0 (probs=1/R)
    cap_sigmoid_v<<<128,  blk, 0, stream>>>(S, v);             // v0, S=0
    cap_logits   <<<256,  blk, 0, stream>>>(x, w, v, logits, 0);
    // iter 1
    cap_softmax  <<<1024, blk, 0, stream>>>(logits, probs);
    cap_gemmS    <<<256,  blk, 0, stream>>>(x, w, probs, S);   // S1
    cap_sigmoid_v<<<128,  blk, 0, stream>>>(S, v);             // v1, S=0
    cap_logits   <<<256,  blk, 0, stream>>>(x, w, v, logits, 1);
    // iter 2
    cap_softmax  <<<1024, blk, 0, stream>>>(logits, probs);
    cap_gemmS    <<<256,  blk, 0, stream>>>(x, w, probs, S);   // S2
    cap_out      <<<128,  blk, 0, stream>>>(S, out);
}
